// BertMoELayer_42691974922302
// MI455X (gfx1250) — matrix operand order repaired and checked by reading, not compiled, not run on hardware
//
#include <hip/hip_runtime.h>
#include <hip/hip_bf16.h>
#include <math.h>

// Problem constants (B=2, S=2048 -> T=4096)
#define T_TOT   4096
#define D_DIM   768
#define F_DIM   3072
#define E_NUM   8
#define TOK_TILE 32

typedef __attribute__((ext_vector_type(16))) __bf16 v16bf;
typedef __attribute__((ext_vector_type(8)))  float  v8f;
typedef __attribute__((ext_vector_type(4)))  unsigned int uint4v;

union ABOp { uint4v q[2]; v16bf v; };

__device__ __forceinline__ unsigned short f2bf(float f) {
  unsigned int u = __float_as_uint(f);
  u += 0x7FFFu + ((u >> 16) & 1u);   // round-to-nearest-even
  return (unsigned short)(u >> 16);
}

__device__ __forceinline__ float gelu_exact(float x) {
  return 0.5f * x * (1.0f + erff(x * 0.70710678118654752440f));
}

// ---------------------------------------------------------------------------
// Gate: logits = x @ gate_w ; softmax ; top-2 (no renorm) -> dense combine[T,E]
// ---------------------------------------------------------------------------
__global__ __launch_bounds__(256) void gate_kernel(
    const float* __restrict__ x, const float* __restrict__ gw,
    float* __restrict__ combine) {
  const int t = blockIdx.x * blockDim.x + threadIdx.x;  // one token per thread
  const float* xr = x + (size_t)t * D_DIM;
  float acc[E_NUM];
  #pragma unroll
  for (int e = 0; e < E_NUM; ++e) acc[e] = 0.0f;
  for (int d = 0; d < D_DIM; ++d) {
    const float xv = xr[d];
    #pragma unroll
    for (int e = 0; e < E_NUM; ++e) acc[e] += xv * gw[d * E_NUM + e];
  }
  float mx = acc[0];
  #pragma unroll
  for (int e = 1; e < E_NUM; ++e) mx = fmaxf(mx, acc[e]);
  float s = 0.0f, p[E_NUM];
  #pragma unroll
  for (int e = 0; e < E_NUM; ++e) { p[e] = __expf(acc[e] - mx); s += p[e]; }
  const float inv = 1.0f / s;
  #pragma unroll
  for (int e = 0; e < E_NUM; ++e) p[e] *= inv;
  int i1 = 0;
  #pragma unroll
  for (int e = 1; e < E_NUM; ++e) if (p[e] > p[i1]) i1 = e;
  int i2 = (i1 == 0) ? 1 : 0;
  #pragma unroll
  for (int e = 0; e < E_NUM; ++e) if (e != i1 && p[e] > p[i2]) i2 = e;
  #pragma unroll
  for (int e = 0; e < E_NUM; ++e)
    combine[(size_t)t * E_NUM + e] = (e == i1 || e == i2) ? p[e] : 0.0f;
}

// ---------------------------------------------------------------------------
// Transpose + fp32->bf16 convert:  in[E][R][C] f32  ->  out[E][C][R] bf16
// ---------------------------------------------------------------------------
__global__ __launch_bounds__(256) void transpose_cvt_kernel(
    const float* __restrict__ in, unsigned short* __restrict__ out,
    int R, int C) {
  __shared__ float tile[32][33];
  const int e = blockIdx.z;
  const float* src = in + (size_t)e * R * C;
  unsigned short* dst = out + (size_t)e * R * C;
  const int c0 = blockIdx.x * 32, r0 = blockIdx.y * 32;
  #pragma unroll
  for (int j = 0; j < 4; ++j) {
    const int r = r0 + threadIdx.y + j * 8;
    tile[threadIdx.y + j * 8][threadIdx.x] = src[(size_t)r * C + c0 + threadIdx.x];
  }
  __syncthreads();
  #pragma unroll
  for (int j = 0; j < 4; ++j) {
    const int c = c0 + threadIdx.y + j * 8;
    dst[(size_t)c * R + r0 + threadIdx.x] = f2bf(tile[threadIdx.x][threadIdx.y + j * 8]);
  }
}

__global__ __launch_bounds__(256) void zero_kernel(float* __restrict__ p, int n) {
  const int i = blockIdx.x * blockDim.x + threadIdx.x;
  if (i < n) p[i] = 0.0f;
}

// ---------------------------------------------------------------------------
// Fused MoE FFN (dense weighted-combine, one expert per blockIdx.y):
//   h  = gelu(x[32,768] @ w_up[e] + b_up[e])   kept bf16 in LDS [32,3072]
//   y  = h @ w_down[e] + b_down[e]
//   out[t,:] += combine[t,e] * y   (atomic, skipped when weight == 0)
// ---------------------------------------------------------------------------
#define XS_STRIDE 776           // 768 + 8 bf16 pad  -> LDS bank shift of 4
#define HS_STRIDE 3080          // 3072 + 8 bf16 pad -> LDS bank shift of 4
#define SMEM_XS   (TOK_TILE * XS_STRIDE * 2)   // 49664 B
#define SMEM_HS   (TOK_TILE * HS_STRIDE * 2)   // 197120 B
#define SMEM_CS   (TOK_TILE * 4)               // 128 B
#define SMEM_TOTAL (SMEM_XS + SMEM_HS + SMEM_CS)  // 246912 B < 320 KB WGP LDS

__global__ __launch_bounds__(256) void moe_ffn_kernel(
    const float* __restrict__ x,
    const unsigned short* __restrict__ w_upT,    // [E][F][D] bf16
    const unsigned short* __restrict__ w_downT,  // [E][D][F] bf16
    const float* __restrict__ b_up, const float* __restrict__ b_down,
    const float* __restrict__ combine, float* __restrict__ out) {
  extern __shared__ char smem[];
  unsigned short* xs = (unsigned short*)smem;
  unsigned short* hs = (unsigned short*)(smem + SMEM_XS);
  float* cs = (float*)(smem + SMEM_XS + SMEM_HS);

  const int tid  = threadIdx.x;
  const int lane = tid & 31;
  const int wave = tid >> 5;
  const int e    = blockIdx.y;
  const int t0   = blockIdx.x * TOK_TILE;
  const int lhalf = lane >> 4;   // 0 | 1 (ISA wave32 operand half)
  const int lmod  = lane & 15;

  // ---- Phase 0: stage x tile fp32 -> bf16 into LDS; stage combine weights
  #pragma unroll 4
  for (int i = 0; i < (TOK_TILE * D_DIM) / 256; ++i) {
    const int lin = tid + i * 256;
    const int r = lin / D_DIM, d = lin % D_DIM;
    xs[r * XS_STRIDE + d] = f2bf(x[(size_t)(t0 + r) * D_DIM + d]);
  }
  if (tid < TOK_TILE) cs[tid] = combine[(size_t)(t0 + tid) * E_NUM + e];
  __syncthreads();

  // ---- Phase 1: h = gelu(x @ w_up + b_up); each wave owns 384 F-columns
  {
    const int fbase = wave * (F_DIM / 8);
    const char* xsb = (const char*)xs;
    for (int ft = 0; ft < 24; ++ft) {
      const int f0 = fbase + ft * 16;
      // B operand: column N=lmod of w_up slice == row (f0+lmod) of w_upT
      const size_t brow = ((size_t)e * F_DIM + f0 + lmod) * D_DIM + (size_t)lhalf * 8;
      v8f acc0 = {}; v8f acc1 = {};
      for (int k = 0; k < 24; ++k) {             // K = 768 in steps of 32
        ABOp b;
        const unsigned short* bp = w_upT + brow + k * 32;
        b.q[0] = *(const uint4v*)(bp);
        b.q[1] = *(const uint4v*)(bp + 16);
        ABOp a0, a1;                              // A rows 0-15 and 16-31
        int off = lmod * (XS_STRIDE * 2) + k * 64 + lhalf * 16;
        a0.q[0] = *(const uint4v*)(xsb + off);
        a0.q[1] = *(const uint4v*)(xsb + off + 32);
        off += 16 * (XS_STRIDE * 2);
        a1.q[0] = *(const uint4v*)(xsb + off);
        a1.q[1] = *(const uint4v*)(xsb + off + 32);
        acc0 = __builtin_amdgcn_wmma_f32_16x16x32_bf16(false, a0.v, false, b.v,
                                                       (short)0, acc0, false, false);
        acc1 = __builtin_amdgcn_wmma_f32_16x16x32_bf16(false, a1.v, false, b.v,
                                                       (short)0, acc1, false, false);
      }
      const int col = f0 + lmod;
      const float bias = b_up[e * F_DIM + col];
      #pragma unroll
      for (int r = 0; r < 8; ++r) {               // C layout: M = r + 8*lhalf
        const int m0 = r + 8 * lhalf;
        hs[m0 * HS_STRIDE + col]        = f2bf(gelu_exact(acc0[r] + bias));
        hs[(m0 + 16) * HS_STRIDE + col] = f2bf(gelu_exact(acc1[r] + bias));
      }
    }
  }
  __syncthreads();

  // ---- Phase 2: y = h @ w_down; each wave owns 96 D-columns (12 acc tiles)
  {
    const int dbase = wave * (D_DIM / 8);
    v8f acc[12];
    #pragma unroll
    for (int i = 0; i < 12; ++i) acc[i] = {};
    const char* hsb = (const char*)hs;
    for (int k = 0; k < 96; ++k) {               // K = 3072 in steps of 32
      ABOp a0, a1;
      int off = lmod * (HS_STRIDE * 2) + k * 64 + lhalf * 16;
      a0.q[0] = *(const uint4v*)(hsb + off);
      a0.q[1] = *(const uint4v*)(hsb + off + 32);
      off += 16 * (HS_STRIDE * 2);
      a1.q[0] = *(const uint4v*)(hsb + off);
      a1.q[1] = *(const uint4v*)(hsb + off + 32);
      #pragma unroll
      for (int ct = 0; ct < 6; ++ct) {
        const size_t brow =
            ((size_t)e * D_DIM + dbase + ct * 16 + lmod) * F_DIM +
            (size_t)k * 32 + (size_t)lhalf * 8;
        ABOp b;
        const unsigned short* bp = w_downT + brow;
        b.q[0] = *(const uint4v*)(bp);
        b.q[1] = *(const uint4v*)(bp + 16);
        acc[ct]     = __builtin_amdgcn_wmma_f32_16x16x32_bf16(false, a0.v, false, b.v,
                                                              (short)0, acc[ct], false, false);
        acc[6 + ct] = __builtin_amdgcn_wmma_f32_16x16x32_bf16(false, a1.v, false, b.v,
                                                              (short)0, acc[6 + ct], false, false);
      }
    }
    // Epilogue: out[t,d] += combine[t,e] * (y + b_down[e,d]); skip zero gates
    #pragma unroll
    for (int ct = 0; ct < 6; ++ct) {
      const int d = dbase + ct * 16 + lmod;
      const float bias = b_down[e * D_DIM + d];
      #pragma unroll
      for (int rt = 0; rt < 2; ++rt) {
        #pragma unroll
        for (int r = 0; r < 8; ++r) {
          const int m = rt * 16 + r + 8 * lhalf;
          const float w = cs[m];
          if (w != 0.0f) {
            atomicAdd(out + (size_t)(t0 + m) * D_DIM + d,
                      w * (acc[rt * 6 + ct][r] + bias));
          }
        }
      }
    }
  }
}

// ---------------------------------------------------------------------------
extern "C" void kernel_launch(void* const* d_in, const int* in_sizes, int n_in,
                              void* d_out, int out_size, void* d_ws, size_t ws_size,
                              hipStream_t stream) {
  const float* x      = (const float*)d_in[0];   // [2,2048,768]
  const float* gate_w = (const float*)d_in[1];   // [768,8]
  const float* w_up   = (const float*)d_in[2];   // [8,768,3072]
  const float* b_up   = (const float*)d_in[3];   // [8,3072]
  const float* w_down = (const float*)d_in[4];   // [8,3072,768]
  const float* b_down = (const float*)d_in[5];   // [8,768]
  float* out = (float*)d_out;

  char* ws = (char*)d_ws;
  float* combine = (float*)ws;                                   // 131072 B
  unsigned short* w_upT   = (unsigned short*)(ws + 131072);      // 37748736 B
  unsigned short* w_downT = (unsigned short*)(ws + 131072 + 37748736ull);

  // 1) gating
  gate_kernel<<<T_TOT / 256, 256, 0, stream>>>(x, gate_w, combine);
  // 2) weight convert+transpose: w_up[E][D][F] -> w_upT[E][F][D] bf16
  transpose_cvt_kernel<<<dim3(F_DIM / 32, D_DIM / 32, E_NUM), dim3(32, 8), 0, stream>>>(
      w_up, w_upT, D_DIM, F_DIM);
  //    w_down[E][F][D] -> w_downT[E][D][F] bf16
  transpose_cvt_kernel<<<dim3(D_DIM / 32, F_DIM / 32, E_NUM), dim3(32, 8), 0, stream>>>(
      w_down, w_downT, F_DIM, D_DIM);
  // 3) zero output (harness poisons d_out; epilogue uses atomics)
  zero_kernel<<<(T_TOT * D_DIM) / 256, 256, 0, stream>>>(out, T_TOT * D_DIM);
  // 4) fused WMMA FFN, 247 KB dynamic LDS per workgroup
  (void)hipFuncSetAttribute((const void*)moe_ffn_kernel,
                            hipFuncAttributeMaxDynamicSharedMemorySize, SMEM_TOTAL);
  moe_ffn_kernel<<<dim3(T_TOT / TOK_TILE, E_NUM), 256, SMEM_TOTAL, stream>>>(
      x, w_upT, w_downT, b_up, b_down, combine, out);
}
